// Attention_14328010900320
// MI455X (gfx1250) — compile-verified, matrix operand order checked by reading, not run on hardware
//
#include <hip/hip_runtime.h>
#include <hip/hip_bf16.h>

typedef __attribute__((ext_vector_type(16))) _Float16 v16h;
typedef __attribute__((ext_vector_type(8)))  float    v8f;

#define D_MODEL 2048
#define SEQ     2048
#define NH      16
#define NKVH    4
#define HD      128
#define KVD     (NKVH*HD)   // 512
#define WINDOW  1024
#define ATT_SCALE 0.08838834764831845f  // 1/sqrt(128)

// ---------------------------------------------------------------------------
// GEMM: C[M,N] = opt_clip( A[M,K] @ W[N,K]^T ), A/W fp32, out fp32 or f16.
// Block tile 128x128, BK=32, 256 threads = 8 waves (4x2), 32x64 per wave.
// ---------------------------------------------------------------------------
#define BM 128
#define BN 128
#define BK 32
#define LDT 40   // LDS row stride in elements (32 + 8 pad), 80B = 16B aligned

__global__ __launch_bounds__(256) void gemm_xWt_kernel(
    const float* __restrict__ A, const float* __restrict__ W,
    float* __restrict__ outF, _Float16* __restrict__ outH,
    int M, int N, int K, float clipv)
{
    __shared__ _Float16 As[BM * LDT];
    __shared__ _Float16 Bs[BN * LDT];

    const int t    = threadIdx.x;
    const int lane = t & 31;
    const int wid  = t >> 5;
    const int wm   = wid >> 1;   // 0..3  (M direction, 32 rows each)
    const int wn   = wid & 1;    // 0..1  (N direction, 64 cols each)
    const int hi   = lane >> 4;  // 0/1
    const int lo   = lane & 15;

    const int m0 = blockIdx.x * BM;
    const int n0 = blockIdx.y * BN;

    v8f c[2][4] = {};

    for (int k0 = 0; k0 < K; k0 += BK) {
        // stage A tile (128x32 fp32 -> f16)
        #pragma unroll
        for (int i = 0; i < 4; ++i) {
            int e = t * 16 + i * 4;
            int r = e >> 5, cc = e & 31;
            const float4 f = *(const float4*)(A + (size_t)(m0 + r) * K + k0 + cc);
            _Float16* d = &As[r * LDT + cc];
            d[0] = (_Float16)f.x; d[1] = (_Float16)f.y;
            d[2] = (_Float16)f.z; d[3] = (_Float16)f.w;
        }
        // stage W tile (128x32 fp32 -> f16)
        #pragma unroll
        for (int i = 0; i < 4; ++i) {
            int e = t * 16 + i * 4;
            int r = e >> 5, cc = e & 31;
            const float4 f = *(const float4*)(W + (size_t)(n0 + r) * K + k0 + cc);
            _Float16* d = &Bs[r * LDT + cc];
            d[0] = (_Float16)f.x; d[1] = (_Float16)f.y;
            d[2] = (_Float16)f.z; d[3] = (_Float16)f.w;
        }
        if (k0 + BK < K)  // hint next A slab toward L2 (global_prefetch_b8)
            __builtin_prefetch(A + (size_t)(m0 + (t >> 1)) * K + k0 + BK, 0, 1);
        __syncthreads();

        v16h a[2], b[4];
        #pragma unroll
        for (int i = 0; i < 2; ++i)
            a[i] = *(const v16h*)&As[(wm * 32 + 16 * i + lo) * LDT + hi * 16];
        #pragma unroll
        for (int j = 0; j < 4; ++j)
            b[j] = *(const v16h*)&Bs[(wn * 64 + 16 * j + lo) * LDT + hi * 16];

        #pragma unroll
        for (int i = 0; i < 2; ++i)
            #pragma unroll
            for (int j = 0; j < 4; ++j)
                c[i][j] = __builtin_amdgcn_wmma_f32_16x16x32_f16(
                    false, a[i], false, b[j], (short)0, c[i][j], false, false);
        __syncthreads();
    }

    // epilogue: C layout row = r + 8*hi inside 16x16 tile, col = lo
    #pragma unroll
    for (int i = 0; i < 2; ++i)
        #pragma unroll
        for (int j = 0; j < 4; ++j) {
            int rbase = m0 + wm * 32 + 16 * i + 8 * hi;
            int col   = n0 + wn * 64 + 16 * j + lo;
            #pragma unroll
            for (int r = 0; r < 8; ++r) {
                float v = c[i][j][r];
                if (clipv > 0.0f) v = fminf(fmaxf(v, -clipv), clipv);
                size_t idx = (size_t)(rbase + r) * N + col;
                if (outH) outH[idx] = (_Float16)v;
                else      outF[idx] = v;
            }
        }
}

// ---------------------------------------------------------------------------
// In-place RMSNorm + RoPE on f16 rows of length 128. grid=(S, nheads), 128 thr.
// ---------------------------------------------------------------------------
__global__ __launch_bounds__(128) void norm_rope_kernel(
    _Float16* __restrict__ qk, const float* __restrict__ nw, int row_stride)
{
    const int s = blockIdx.x, h = blockIdx.y, i = threadIdx.x;
    _Float16* row = qk + (size_t)s * row_stride + h * HD;

    float x  = (float)row[i];
    float ss = x * x;
    #pragma unroll
    for (int m = 16; m >= 1; m >>= 1) ss += __shfl_xor(ss, m, 32);

    __shared__ float wss[4];
    __shared__ float xn_sh[HD];
    if ((i & 31) == 0) wss[i >> 5] = ss;
    __syncthreads();
    float tot  = wss[0] + wss[1] + wss[2] + wss[3];
    float rstd = rsqrtf(tot * (1.0f / HD) + 1e-6f);
    float xn   = x * rstd * nw[i];
    xn_sh[i] = xn;
    __syncthreads();

    float part = (i < 64) ? -xn_sh[i + 64] : xn_sh[i - 64];
    int   j    = i & 63;
    float ang  = (float)s * __powf(500000.0f, -(float)j * (1.0f / 64.0f));
    float cs, sn;
    __sincosf(ang, &sn, &cs);
    row[i] = (_Float16)(xn * cs + part * sn);
}

// ---------------------------------------------------------------------------
// Sliding-window attention with sink, flash-style online softmax.
// grid=(S/64, H), 128 threads = 4 waves; each wave owns 16 query rows.
// K tile staged with GLOBAL_LOAD_ASYNC_TO_LDS_B128 (ASYNCcnt path).
// ---------------------------------------------------------------------------
#define KLDS 136   // 128 + 8 pad
#define VLDS 72    // 64 + 8 pad
#define PLDS 72

__global__ __launch_bounds__(128) void attn_kernel(
    const _Float16* __restrict__ Q, const _Float16* __restrict__ Km,
    const _Float16* __restrict__ V, const float* __restrict__ sinks,
    float* __restrict__ att)
{
    __shared__ _Float16 Ks[64 * KLDS];        // K tile, [key][hd]
    __shared__ _Float16 Vs[HD * VLDS];        // V tile transposed, [hd][key]
    __shared__ _Float16 Ps[4][16 * PLDS];     // per-wave P scratch

    const int qb = blockIdx.x, h = blockIdx.y;
    const int kvh  = h >> 2;                  // N_REP = 4
    const int t    = threadIdx.x;
    const int lane = t & 31;
    const int wid  = t >> 5;
    const int hi   = lane >> 4;
    const int lo   = lane & 15;
    const int q0   = qb * 64;
    const int qrow = q0 + wid * 16 + lo;      // A-fragment row for this lane

    // Q fragments (16x128 per wave -> 4 frags of 16x32)
    v16h qf[4];
    #pragma unroll
    for (int f = 0; f < 4; ++f)
        qf[f] = *(const v16h*)(Q + (size_t)qrow * D_MODEL + h * HD + f * 32 + hi * 16);

    v8f   o[8] = {};
    float m[8], l[8];
    const float sink = sinks[h];
    #pragma unroll
    for (int r = 0; r < 8; ++r) { m[r] = sink; l[r] = 1.0f; }

    int lo_k  = q0 - (WINDOW - 1); if (lo_k < 0) lo_k = 0;
    const int kt_lo = (lo_k >> 6) << 6;
    const int myrow = q0 + wid * 16 + 8 * hi;  // C-layout row base for this lane

    for (int kt = kt_lo; kt <= q0 + 63; kt += 64) {
        __syncthreads();
        { // K tile: 64 rows x 256B, async DMA straight into LDS, 16B per lane
            #pragma unroll
            for (int i = 0; i < 8; ++i) {
                int b    = i * 2048 + t * 16;     // byte index in 16 KB tile
                int row  = b >> 8;                // 256B per key row
                int colb = b & 255;
                const char* ga =
                    (const char*)(Km + (size_t)(kt + row) * KVD + kvh * HD) + colb;
                unsigned lds = (unsigned)(size_t)&Ks[row * KLDS] + colb;
                asm volatile("global_load_async_to_lds_b128 %0, %1, off"
                             :: "v"(lds), "v"((unsigned long long)(size_t)ga)
                             : "memory");
            }
        }
        { // V tile with transpose: thread t = head-dim column
            #pragma unroll 4
            for (int key = 0; key < 64; ++key)
                Vs[t * VLDS + key] = V[(size_t)(kt + key) * KVD + kvh * HD + t];
        }
        asm volatile("s_wait_asynccnt 0x0" ::: "memory");
        __syncthreads();

        // S = Q @ K^T  (4 tiles of 16x16 along keys)
        v8f st[4] = {};
        #pragma unroll
        for (int j = 0; j < 4; ++j)
            #pragma unroll
            for (int f = 0; f < 4; ++f) {
                v16h kb = *(const v16h*)&Ks[(16 * j + lo) * KLDS + f * 32 + hi * 16];
                st[j] = __builtin_amdgcn_wmma_f32_16x16x32_f16(
                    false, qf[f], false, kb, (short)0, st[j], false, false);
            }

        // mask + scale, row-max
        float rowm[8];
        #pragma unroll
        for (int r = 0; r < 8; ++r) rowm[r] = -3.0e38f;
        #pragma unroll
        for (int j = 0; j < 4; ++j) {
            int col = kt + 16 * j + lo;
            #pragma unroll
            for (int r = 0; r < 8; ++r) {
                int  qr = myrow + r;
                bool ok = (col <= qr) && (col > qr - WINDOW);
                float sv = ok ? st[j][r] * ATT_SCALE : -3.0e38f;
                st[j][r] = sv;
                rowm[r]  = fmaxf(rowm[r], sv);
            }
        }
        #pragma unroll
        for (int r = 0; r < 8; ++r)
            #pragma unroll
            for (int msk = 8; msk >= 1; msk >>= 1)
                rowm[r] = fmaxf(rowm[r], __shfl_xor(rowm[r], msk, 16));

        // online softmax update
        float alpha[8], mnew[8], rsum[8];
        #pragma unroll
        for (int r = 0; r < 8; ++r) {
            mnew[r]  = fmaxf(m[r], rowm[r]);
            alpha[r] = __expf(m[r] - mnew[r]);
            rsum[r]  = 0.0f;
        }
        #pragma unroll
        for (int j = 0; j < 4; ++j)
            #pragma unroll
            for (int r = 0; r < 8; ++r) {
                float p = __expf(st[j][r] - mnew[r]);
                st[j][r] = p;
                rsum[r] += p;
            }
        #pragma unroll
        for (int r = 0; r < 8; ++r) {
            #pragma unroll
            for (int msk = 8; msk >= 1; msk >>= 1)
                rsum[r] += __shfl_xor(rsum[r], msk, 16);
            l[r] = l[r] * alpha[r] + rsum[r];
            m[r] = mnew[r];
        }
        #pragma unroll
        for (int n = 0; n < 8; ++n)
            #pragma unroll
            for (int r = 0; r < 8; ++r) o[n][r] *= alpha[r];

        // C-layout P -> per-wave LDS -> A fragments (same-wave DS is in-order)
        #pragma unroll
        for (int j = 0; j < 4; ++j)
            #pragma unroll
            for (int r = 0; r < 8; ++r)
                Ps[wid][(r + 8 * hi) * PLDS + 16 * j + lo] = (_Float16)st[j][r];

        // O += P @ V
        #pragma unroll
        for (int ks = 0; ks < 2; ++ks) {
            v16h pa = *(const v16h*)&Ps[wid][lo * PLDS + ks * 32 + hi * 16];
            #pragma unroll
            for (int n = 0; n < 8; ++n) {
                v16h vb = *(const v16h*)&Vs[(16 * n + lo) * VLDS + ks * 32 + hi * 16];
                o[n] = __builtin_amdgcn_wmma_f32_16x16x32_f16(
                    false, pa, false, vb, (short)0, o[n], false, false);
            }
        }
    }

    // finalize: O / l  ->  att[s][h*128 + hd] (fp32)
    #pragma unroll
    for (int n = 0; n < 8; ++n) {
        #pragma unroll
        for (int r = 0; r < 8; ++r) {
            int row = myrow + r;
            att[(size_t)row * D_MODEL + h * HD + 16 * n + lo] = o[n][r] / l[r];
        }
    }
}

// ---------------------------------------------------------------------------
extern "C" void kernel_launch(void* const* d_in, const int* in_sizes, int n_in,
                              void* d_out, int out_size, void* d_ws, size_t ws_size,
                              hipStream_t stream)
{
    const float* x      = (const float*)d_in[0];
    const float* w_q    = (const float*)d_in[1];
    const float* w_k    = (const float*)d_in[2];
    const float* w_v    = (const float*)d_in[3];
    const float* w_out  = (const float*)d_in[4];
    const float* q_nw   = (const float*)d_in[5];
    const float* k_nw   = (const float*)d_in[6];
    const float* sinks  = (const float*)d_in[7];
    float* out = (float*)d_out;

    // workspace layout (28 MB): q f16 | k f16 | v f16 | att f32
    _Float16* qbuf = (_Float16*)d_ws;
    _Float16* kbuf = qbuf + (size_t)SEQ * D_MODEL;
    _Float16* vbuf = kbuf + (size_t)SEQ * KVD;
    float*    attb = (float*)(vbuf + (size_t)SEQ * KVD);

    dim3 blk(256);
    // Q/K/V projections with clip, f16 out
    gemm_xWt_kernel<<<dim3(SEQ / BM, D_MODEL / BN), blk, 0, stream>>>(
        x, w_q, nullptr, qbuf, SEQ, D_MODEL, D_MODEL, 8.0f);
    gemm_xWt_kernel<<<dim3(SEQ / BM, KVD / BN), blk, 0, stream>>>(
        x, w_k, nullptr, kbuf, SEQ, KVD, D_MODEL, 8.0f);
    gemm_xWt_kernel<<<dim3(SEQ / BM, KVD / BN), blk, 0, stream>>>(
        x, w_v, nullptr, vbuf, SEQ, KVD, D_MODEL, 8.0f);

    // RMSNorm + RoPE in place
    norm_rope_kernel<<<dim3(SEQ, NH),   128, 0, stream>>>(qbuf, q_nw, D_MODEL);
    norm_rope_kernel<<<dim3(SEQ, NKVH), 128, 0, stream>>>(kbuf, k_nw, KVD);

    // attention
    attn_kernel<<<dim3(SEQ / 64, NH), 128, 0, stream>>>(qbuf, kbuf, vbuf, sinks, attb);

    // output projection, fp32 out, no clip
    gemm_xWt_kernel<<<dim3(SEQ / BM, D_MODEL / BN), blk, 0, stream>>>(
        attb, w_out, out, nullptr, SEQ, D_MODEL, D_MODEL, 0.0f);
}
